// QKVAttention_17197049053392
// MI455X (gfx1250) — matrix operand order repaired and checked by reading, not compiled, not run on hardware
//
#include <hip/hip_runtime.h>
#include <stdint.h>

typedef __attribute__((ext_vector_type(16))) _Float16 v16h;
typedef __attribute__((ext_vector_type(8)))  _Float16 v8h;
typedef __attribute__((ext_vector_type(8)))  float    v8f;

namespace {
constexpr int   kHeads  = 16;
constexpr int   kCh     = 64;
constexpr int   kT      = 2048;
constexpr int   kS      = 512;
constexpr int   kWaves  = 8;
constexpr int   kTBlk   = 16 * kWaves;          // 128 t-columns per workgroup
constexpr int   kCW     = 64;                   // s-columns per chunk
constexpr int   kChunks = (kS + kT) / kCW;      // 40 chunks
constexpr int   kEncChunks = kS / kCW;          // 8 encoder chunks
constexpr int   kKStr   = 72;                   // K LDS row stride (halves): 36*lane bank starts
constexpr int   kVStr   = 72;                   // V LDS row stride (halves): same pattern
constexpr float kScale  = 0.35355339059327379f; // 64^-0.25 (K side)
constexpr float kLog2e  = 1.4426950408889634f;
constexpr float kScaleQ = kScale * kLog2e;      // Q side: fold log2e into the GEMM
}

union HalfPack { uint32_t u[8]; v16h v; };

__device__ __forceinline__ uint32_t pack2(float lo, float hi) {
    union { _Float16 h[2]; uint32_t u; } x;
    x.h[0] = (_Float16)lo; x.h[1] = (_Float16)hi;
    return x.u;
}

__device__ __forceinline__ v16h cat8(v8h lo, v8h hi) {
    return __builtin_shufflevector(lo, hi, 0, 1, 2, 3, 4, 5, 6, 7,
                                           8, 9, 10, 11, 12, 13, 14, 15);
}

struct Pref { float4 k[4]; float4 v[4]; };

__global__ __launch_bounds__(32 * kWaves)
void qkv_attn_wmma(const float* __restrict__ qkv,
                   const float* __restrict__ enc,
                   float* __restrict__ out)
{
    __shared__ __align__(16) _Float16 kTile[2][kCW * kKStr];  // [s][c] f16 (transposed)
    __shared__ __align__(16) _Float16 vTile[2][kCh * kVStr];  // [c][s] f16

    const int tid    = threadIdx.x;
    const int lane   = tid & 31;
    const int lane16 = lane & 15;
    const int halfid = lane >> 4;
    const int wave   = tid >> 5;

    const int bh = blockIdx.y;              // b*16 + h
    const int b  = bh >> 4;
    const int h  = bh & 15;
    const int t  = blockIdx.x * kTBlk + wave * 16 + lane16;

    const float* qB = qkv + (size_t)b * (3 * kHeads * kCh) * kT + (size_t)(h * kCh) * kT;
    const float* kM = qB + (size_t)(kHeads * kCh) * kT;
    const float* vM = qB + (size_t)(2 * kHeads * kCh) * kT;
    const float* kE = enc + (size_t)b * (2 * kHeads * kCh) * kS + (size_t)(h * kCh) * kS;
    const float* vE = kE + (size_t)(kHeads * kCh) * kS;

    // staging split: workgroup covers one 64c x 64s K tile + V tile per chunk
    const int kpair = lane;                 // K: c pair = 2*lane
    const int ksrow = wave * 8;             // K: 8 consecutive s rows per wave
    const int vc    = tid >> 2;             // V: channel row 0..63
    const int vs    = (tid & 3) * 16;       // V: 16 consecutive s per thread

    auto loadChunk = [&](int i) -> Pref {
        const float *kp, *vp; int stride, s0;
        if (i < kEncChunks) { kp = kE; vp = vE; stride = kS; s0 = i * kCW; }
        else                { kp = kM; vp = vM; stride = kT; s0 = (i - kEncChunks) * kCW; }
        Pref c;
        const float* kr = kp + (size_t)(2 * kpair) * stride + s0 + ksrow;
        c.k[0] = *(const float4*)kr;
        c.k[1] = *(const float4*)(kr + 4);
        c.k[2] = *(const float4*)(kr + stride);
        c.k[3] = *(const float4*)(kr + stride + 4);
        const float* vr = vp + (size_t)vc * stride + s0 + vs;
        c.v[0] = *(const float4*)vr;
        c.v[1] = *(const float4*)(vr + 4);
        c.v[2] = *(const float4*)(vr + 8);
        c.v[3] = *(const float4*)(vr + 12);
        return c;
    };

    auto storeChunk = [&](const Pref& c, int buf) {
        _Float16* kd = &kTile[buf][0];
        const float* r0 = (const float*)&c.k[0];   // row c=2p,   s = ksrow..ksrow+7
        const float* r1 = (const float*)&c.k[2];   // row c=2p+1
        #pragma unroll
        for (int j = 0; j < 8; ++j)
            *(uint32_t*)&kd[(ksrow + j) * kKStr + 2 * kpair] =
                pack2(r0[j] * kScale, r1[j] * kScale);
        const float* vv = (const float*)&c.v[0];
        uint32_t w[8];
        #pragma unroll
        for (int j = 0; j < 8; ++j) w[j] = pack2(vv[2 * j], vv[2 * j + 1]);
        *(uint4*)&vTile[buf][vc * kVStr + vs]     = *(const uint4*)&w[0];
        *(uint4*)&vTile[buf][vc * kVStr + vs + 8] = *(const uint4*)&w[4];
    };

    // ---- resident Q B-fragments (scaled by 64^-0.25 * log2e) ----
    v16h qb0, qb1;
    #pragma unroll
    for (int e = 0; e < 16; ++e) {
        const int c = (e & 7) + 8 * halfid + 16 * (e >> 3);
        qb0[e] = (_Float16)(qB[(size_t)c        * kT + t] * kScaleQ);
        qb1[e] = (_Float16)(qB[(size_t)(c + 32) * kT + t] * kScaleQ);
    }

    v8f acc[4] = {v8f{}, v8f{}, v8f{}, v8f{}};
    float mrun = -3.0e38f, lrun = 0.0f;

    auto consume = [&](int buf) {
        const _Float16* kt = &kTile[buf][0];
        const _Float16* vt = &vTile[buf][0];

        // 4 score tiles S'[s][t] (16 s-rows each), K-dim = ch = 64
        v8f st[4];
        #pragma unroll
        for (int u = 0; u < 4; ++u) {
            const int kb = (u * 16 + lane16) * kKStr + 8 * halfid;
            const v16h f0 = cat8(*(const v8h*)&kt[kb],      *(const v8h*)&kt[kb + 16]);
            const v16h f1 = cat8(*(const v8h*)&kt[kb + 32], *(const v8h*)&kt[kb + 48]);
            v8f z = {};
            v8f s = __builtin_amdgcn_wmma_f32_16x16x32_f16(false, f0, false, qb0, (short)0, z, false, false);
            st[u] = __builtin_amdgcn_wmma_f32_16x16x32_f16(false, f1, false, qb1, (short)0, s, false, false);
        }

        // online softmax over 64 columns (scores already in log2 domain)
        float mx = st[0][0];
        #pragma unroll
        for (int u = 0; u < 4; ++u)
            #pragma unroll
            for (int r = 0; r < 8; ++r) mx = fmaxf(mx, st[u][r]);
        mx = fmaxf(mx, __shfl_xor(mx, 16, 32));
        const float mnew  = fmaxf(mrun, mx);
        const float alpha = __builtin_amdgcn_exp2f(mrun - mnew);

        float p[4][8], sum = 0.0f;
        #pragma unroll
        for (int u = 0; u < 4; ++u)
            #pragma unroll
            for (int r = 0; r < 8; ++r) {
                p[u][r] = __builtin_amdgcn_exp2f(st[u][r] - mnew);
                sum += p[u][r];
            }
        sum += __shfl_xor(sum, 16, 32);
        lrun = lrun * alpha + sum;
        mrun = mnew;

        #pragma unroll
        for (int ct = 0; ct < 4; ++ct)
            #pragma unroll
            for (int r = 0; r < 8; ++r) acc[ct][r] *= alpha;

        // two P B-fragments (s 0..31 from tiles 0,1; s 32..63 from tiles 2,3)
        HalfPack pf[2];
        #pragma unroll
        for (int g = 0; g < 2; ++g) {
            uint32_t olo[4], ohi[4];
            #pragma unroll
            for (int r = 0; r < 4; ++r) {
                olo[r] = pack2(p[2 * g][2 * r],     p[2 * g][2 * r + 1]);
                ohi[r] = pack2(p[2 * g + 1][2 * r], p[2 * g + 1][2 * r + 1]);
            }
            #pragma unroll
            for (int r = 0; r < 4; ++r) {
                const uint32_t xlo = __shfl_xor(olo[r], 16, 32);
                const uint32_t xhi = __shfl_xor(ohi[r], 16, 32);
                pf[g].u[r]     = olo[r];
                pf[g].u[r + 4] = ohi[r];
            }
        }

        // PV: 4 c-tiles x 2 s-subchunks
        #pragma unroll
        for (int ct = 0; ct < 4; ++ct) {
            const int vb = (ct * 16 + lane16) * kVStr + 8 * halfid;
            const v16h va0 = cat8(*(const v8h*)&vt[vb],      *(const v8h*)&vt[vb + 16]);
            const v16h va1 = cat8(*(const v8h*)&vt[vb + 32], *(const v8h*)&vt[vb + 48]);
            acc[ct] = __builtin_amdgcn_wmma_f32_16x16x32_f16(false, va0, false, pf[0].v, (short)0, acc[ct], false, false);
            acc[ct] = __builtin_amdgcn_wmma_f32_16x16x32_f16(false, va1, false, pf[1].v, (short)0, acc[ct], false, false);
        }
    };

    // ---- double-buffered pipeline ----
    Pref c0 = loadChunk(0);
    storeChunk(c0, 0);
    __syncthreads();

    for (int i = 0; i < kChunks; ++i) {
        const int cb = i & 1;
        Pref cn = loadChunk(i < kChunks - 1 ? i + 1 : kChunks - 1);
        consume(cb);
        if (i < kChunks - 1) storeChunk(cn, cb ^ 1);
        __syncthreads();
    }

    // ---- epilogue: out'[c][t]; lane16 = t, VGPR r -> c = r + 8*halfid ----
    const float inv = 1.0f / lrun;
    float* ob = out + (size_t)b * (kHeads * kCh) * kT + (size_t)(h * kCh) * kT;
    #pragma unroll
    for (int ct = 0; ct < 4; ++ct)
        #pragma unroll
        for (int r = 0; r < 8; ++r) {
            const int c = ct * 16 + r + 8 * halfid;
            ob[(size_t)c * kT + t] = acc[ct][r] * inv;
        }
}

extern "C" void kernel_launch(void* const* d_in, const int* in_sizes, int n_in,
                              void* d_out, int out_size, void* d_ws, size_t ws_size,
                              hipStream_t stream) {
    const float* qkv = (const float*)d_in[0];   // [4, 3*16*64, 2048] f32
    const float* enc = (const float*)d_in[1];   // [4, 2*16*64, 512]  f32
    float* out = (float*)d_out;                 // [4, 16*64, 2048]   f32

    dim3 grid(kT / kTBlk, 4 * kHeads, 1);       // (16, 64)
    dim3 block(32 * kWaves, 1, 1);              // 256 threads = 8 wave32 waves
    qkv_attn_wmma<<<grid, block, 0, stream>>>(qkv, enc, out);
}
